// EagerBidirectionalAttentionBlock_56212531970065
// MI455X (gfx1250) — matrix operand order repaired and checked by reading, not compiled, not run on hardware
//
#include <hip/hip_runtime.h>
#include <hip/hip_bf16.h>

// ---------------- problem constants ----------------
#define BATCH 4
#define SEQ   2048
#define DMODEL 1024
#define HEADS 16
#define HEAD_DIM 64
#define MROWS (BATCH * SEQ)   // 8192

typedef __attribute__((ext_vector_type(16))) __bf16 v16bf;
typedef __attribute__((ext_vector_type(8)))  __bf16 v8bf;
typedef __attribute__((ext_vector_type(8)))  float  v8f;
typedef __attribute__((ext_vector_type(4)))  unsigned int u32x4;
typedef __attribute__((ext_vector_type(8)))  int  i32x8;
typedef __attribute__((ext_vector_type(4)))  int  i32x4;

static __device__ __forceinline__ v8f wmma_bf16(v16bf a, v16bf b, v8f c) {
    // D = A(16x32 bf16) * B(32x16 bf16) + C(16x16 f32)
    return __builtin_amdgcn_wmma_f32_16x16x32_bf16(
        /*neg_a=*/false, a, /*neg_b=*/false, b,
        /*c_mod=*/(short)0, c, /*reuse_a=*/false, /*reuse_b=*/false);
}

// Build a 16-element bf16 fragment from two contiguous 16-byte chunks.
static __device__ __forceinline__ v16bf ld2x8(const __bf16* plo, const __bf16* phi) {
    v8bf lo = *(const v8bf*)plo;
    v8bf hi = *(const v8bf*)phi;
    return __builtin_shufflevector(lo, hi, 0,1,2,3,4,5,6,7,8,9,10,11,12,13,14,15);
}

// ---- Tensor Data Mover: 2D tile load (bf16 elements) global -> LDS ----
// Tile is tile1 rows x 64 elements (128B); TDM pads +16B after every 128B row
// so LDS row stride is 144B (36 banks -> <=2-way ds_load_b128 conflicts).
// Descriptor bit layout per CDNA5 ISA 8.3/8.4 (D# group0/group1, 2D, no iterate).
static __device__ __forceinline__ void tdm_load_tile(
        unsigned lds_addr, const void* gaddr,
        unsigned dim0, unsigned dim1, unsigned stride0, unsigned tile1) {
    unsigned long long ga = (unsigned long long)gaddr;
    u32x4 g0 = { 1u,                                   // count = 1 valid descriptor
                 lds_addr,                             // LDS byte address
                 (unsigned)ga,                         // global addr [31:0]
                 (unsigned)(ga >> 32) | (2u << 30) };  // global addr [56:32], type=2
    i32x8 g1 = { (int)((1u << 16)        // data_size = 1 -> 2-byte elements
                     | (1u << 20)        // pad_enable
                     | (4u << 22)        // pad_interval: code 4 -> 32 DWORDs (128B)
                     | (3u << 25)),      // pad_amount:   code 3 -> 4 DWORDs (16B)
                 (int)((dim0 & 0xffffu) << 16),                 // tensor_dim0[15:0]
                 (int)((dim0 >> 16) | ((dim1 & 0xffffu) << 16)),// dim0 hi | dim1 lo
                 (int)((dim1 >> 16) | (64u << 16)),             // dim1 hi | tile_dim0=64
                 (int)tile1,                                    // tile_dim1 | tile_dim2=0
                 (int)stride0,                                  // tensor_dim0_stride lo
                 0, 0 };
    i32x4 z4 = {0, 0, 0, 0};
#if defined(__clang_major__) && __clang_major__ >= 23
    i32x8 z8 = {0, 0, 0, 0, 0, 0, 0, 0};
    __builtin_amdgcn_tensor_load_to_lds(g0, g1, z4, z4, z8, 0);
#else
    __builtin_amdgcn_tensor_load_to_lds(g0, g1, z4, z4, 0);
#endif
}

// ---------------- precision conversion ----------------
__global__ void cvt_x_kernel(const float* __restrict__ x, __bf16* __restrict__ xb, int n) {
    int i = blockIdx.x * blockDim.x + threadIdx.x;
    if (i < n) xb[i] = (__bf16)x[i];
}

// Transpose-convert each weight: W[k][n] (fp32) -> Wt[n][k] (bf16), z selects matrix.
__global__ void cvt_wt_kernel(const float* __restrict__ Wq, const float* __restrict__ Wk,
                              const float* __restrict__ Wv, const float* __restrict__ Wo,
                              __bf16* __restrict__ Wt) {
    int z = blockIdx.z;
    const float* w = (z == 0) ? Wq : (z == 1) ? Wk : (z == 2) ? Wv : Wo;
    int i = blockIdx.x * blockDim.x + threadIdx.x;   // output-linear index n*D + k
    int n = i >> 10;            // / DMODEL
    int k = i & (DMODEL - 1);
    Wt[(size_t)z * DMODEL * DMODEL + i] = (__bf16)w[(size_t)k * DMODEL + n];
}

// ---------------- QKV projection GEMM ----------------
// C[M=8192, N=1024] = X[M,K=1024] * W + bias, bf16 WMMA, fp32 accum.
// grid: (N/128, M/128, 3)  block: 128 threads = 4 waves, wave tile 64x64.
__global__ __launch_bounds__(128, 1)
void qkv_gemm_kernel(const __bf16* __restrict__ X, const __bf16* __restrict__ WtAll,
                     const float* __restrict__ bq, const float* __restrict__ bk,
                     const float* __restrict__ bv,
                     __bf16* __restrict__ Qo, __bf16* __restrict__ Ko,
                     __bf16* __restrict__ Vt) {
    const int z    = blockIdx.z;
    const __bf16* W = WtAll + (size_t)z * DMODEL * DMODEL;   // [N,K] row-major
    const float* bias = (z == 0) ? bq : (z == 1) ? bk : bv;

    const int lane = threadIdx.x & 31;
    const int wid  = threadIdx.x >> 5;
    const int kh   = lane >> 4;      // half-wave (K-split within fragments)
    const int l16  = lane & 15;
    const int m0 = blockIdx.y * 128 + (wid >> 1) * 64;
    const int n0 = blockIdx.x * 128 + (wid & 1) * 64;

    const __bf16* Abase = X + (size_t)(m0 + l16) * DMODEL + kh * 8;
    const __bf16* Bbase = W + (size_t)(n0 + l16) * DMODEL + kh * 8;

    v8f acc[4][4] = {};
    for (int kb = 0; kb < DMODEL; kb += 32) {
        v16bf afr[4];
        #pragma unroll
        for (int i = 0; i < 4; ++i) {
            const __bf16* p = Abase + i * (16 * DMODEL);     // +32KB*i immediate
            afr[i] = ld2x8(p, p + 16);                       // K chunks kh*8, 16+kh*8
        }
        __builtin_prefetch(Abase + 64, 0, 1);                // next-next K tile
        #pragma unroll
        for (int j = 0; j < 4; ++j) {
            const __bf16* p = Bbase + j * (16 * DMODEL);     // +32KB*j immediate
            v16bf bfr = ld2x8(p, p + 16);                    // K chunks kh*8, 16+kh*8
            #pragma unroll
            for (int i = 0; i < 4; ++i)
                acc[i][j] = wmma_bf16(afr[i], bfr, acc[i][j]);
        }
        Abase += 32;
        Bbase += 32;
    }

    // Epilogue: add bias, scatter to attention-friendly layouts.
    #pragma unroll
    for (int i = 0; i < 4; ++i) {
        #pragma unroll
        for (int j = 0; j < 4; ++j) {
            const int n  = n0 + j * 16 + l16;
            const float bn = bias[n];
            const int h  = n >> 6;          // / HEAD_DIM
            const int dd = n & 63;
            #pragma unroll
            for (int r = 0; r < 8; ++r) {
                const int m = m0 + i * 16 + r + kh * 8;
                const int b = m >> 11;      // / SEQ
                const int s = m & (SEQ - 1);
                const __bf16 o = (__bf16)(acc[i][j][r] + bn);
                const size_t bh = (size_t)b * HEADS + h;
                if (z == 0)      Qo[(bh * SEQ + s) * HEAD_DIM + dd] = o;  // [bh,S,dh]
                else if (z == 1) Ko[(bh * SEQ + s) * HEAD_DIM + dd] = o;  // [bh,S,dh]
                else             Vt[(bh * HEAD_DIM + dd) * SEQ + s] = o;  // [bh,dh,S]
            }
        }
    }
}

// ---------------- flash attention (TDM double-buffered K/V in LDS) ----------------
// grid: (S/64, B*H)  block: 128 = 4 waves; each wave owns 16 query rows.
#define KV_ROW_B 144                      // 128B row + 16B TDM pad
#define KV_TILE_B (64 * KV_ROW_B)         // one 64-row tile in LDS

__global__ __launch_bounds__(128, 1)
void attn_kernel(const __bf16* __restrict__ Q, const __bf16* __restrict__ K,
                 const __bf16* __restrict__ V,  // V transposed [bh,dh,S]
                 const unsigned char* __restrict__ mask,
                 __bf16* __restrict__ Ob) {
    __shared__ __attribute__((aligned(16))) char kbuf[2][KV_TILE_B];
    __shared__ __attribute__((aligned(16))) char vbuf[2][KV_TILE_B];
    __shared__ __bf16 pbuf[4][16][80];    // 160B row stride, 16B aligned

    const int lane = threadIdx.x & 31;
    const int wid  = threadIdx.x >> 5;
    const int kh   = lane >> 4;
    const int l16  = lane & 15;
    const int bh = blockIdx.y;
    const int b  = bh >> 4;
    const int h  = bh & 15;
    const int q0 = blockIdx.x * 64 + wid * 16;

    const __bf16* Qp = Q + (size_t)bh * SEQ * HEAD_DIM;
    const __bf16* Kp = K + (size_t)bh * SEQ * HEAD_DIM;
    const __bf16* Vp = V + (size_t)bh * HEAD_DIM * SEQ;
    const unsigned char* mbase = mask + (size_t)b * SEQ + l16;

    // Q fragments (held across the whole KV loop): dh = 64 -> 2 K-steps of 32.
    v16bf aq[2];
    {
        const __bf16* p = Qp + (size_t)(q0 + l16) * HEAD_DIM + kh * 8;
        aq[0] = ld2x8(p,      p + 16);
        aq[1] = ld2x8(p + 32, p + 48);
    }

    // Prologue: wave 0 issues TDM loads of KV block 0 into buffer 0.
    if (wid == 0) {
        tdm_load_tile((unsigned)(size_t)&kbuf[0][0], Kp,
                      HEAD_DIM, SEQ, HEAD_DIM, 64);   // K rows: 64 x dh
        tdm_load_tile((unsigned)(size_t)&vbuf[0][0], Vp,
                      SEQ, HEAD_DIM, SEQ, 64);        // V^T rows: dh x 64 cols
    }

    float mrow[8], lrow[8];
    v8f accO[4] = {};
    #pragma unroll
    for (int r = 0; r < 8; ++r) { mrow[r] = -1e30f; lrow[r] = 0.f; }

    const int NBLK = SEQ / 64;
    for (int ib = 0; ib < NBLK; ++ib) {
        const int cur = ib & 1;
        if (wid == 0) __builtin_amdgcn_s_wait_tensorcnt(0);   // current tiles landed
        __syncthreads();                                      // publish LDS to all waves
        if (wid == 0 && ib + 1 < NBLK) {                      // overlap next DMA w/ math
            tdm_load_tile((unsigned)(size_t)&kbuf[cur ^ 1][0],
                          Kp + (size_t)(ib + 1) * 64 * HEAD_DIM,
                          HEAD_DIM, SEQ, HEAD_DIM, 64);
            tdm_load_tile((unsigned)(size_t)&vbuf[cur ^ 1][0],
                          Vp + (size_t)(ib + 1) * 64,
                          SEQ, HEAD_DIM, SEQ, 64);
        }
        const char* kt = &kbuf[cur][0];
        const char* vt = &vbuf[cur][0];

        // ---- scores: S = Q * K^T / sqrt(dh), 64 KV columns ----
        v8f sc[4];
        #pragma unroll
        for (int j = 0; j < 4; ++j) {
            const char* kr = kt + (j * 16 + l16) * KV_ROW_B + kh * 16;
            v16bf b0 = ld2x8((const __bf16*)kr,        (const __bf16*)(kr + 32));
            v16bf b1 = ld2x8((const __bf16*)(kr + 64), (const __bf16*)(kr + 96));
            v8f s = {};
            s = wmma_bf16(aq[0], b0, s);
            s = wmma_bf16(aq[1], b1, s);
            const bool msk = mbase[j * 16] != 0;             // column masked for every row
            #pragma unroll
            for (int r = 0; r < 8; ++r)
                sc[j][r] = msk ? -1e30f : s[r] * 0.125f;     // 1/sqrt(64)
        }
        // ---- online softmax (row = r + 8*kh, 16 lanes of a half hold the cols) ----
        #pragma unroll
        for (int r = 0; r < 8; ++r) {
            float v = fmaxf(fmaxf(sc[0][r], sc[1][r]), fmaxf(sc[2][r], sc[3][r]));
            v = fmaxf(v, __shfl_xor(v, 1, 32));
            v = fmaxf(v, __shfl_xor(v, 2, 32));
            v = fmaxf(v, __shfl_xor(v, 4, 32));
            v = fmaxf(v, __shfl_xor(v, 8, 32));
            const float mnew = fmaxf(mrow[r], v);
            const float c = __expf(mrow[r] - mnew);
            mrow[r] = mnew;
            lrow[r] *= c;
            #pragma unroll
            for (int jd = 0; jd < 4; ++jd) accO[jd][r] *= c;
        }
        #pragma unroll
        for (int j = 0; j < 4; ++j) {
            #pragma unroll
            for (int r = 0; r < 8; ++r) {
                const float p = __expf(sc[j][r] - mrow[r]);
                sc[j][r] = p;
                pbuf[wid][r + kh * 8][j * 16 + l16] = (__bf16)p;  // C-layout -> row-major
            }
        }
        #pragma unroll
        for (int r = 0; r < 8; ++r) {
            float v = sc[0][r] + sc[1][r] + sc[2][r] + sc[3][r];
            v += __shfl_xor(v, 1, 32);
            v += __shfl_xor(v, 2, 32);
            v += __shfl_xor(v, 4, 32);
            v += __shfl_xor(v, 8, 32);
            lrow[r] += v;
        }
        __syncthreads();
        // ---- reload P as A-fragments (ds_load_b128) ----
        v16bf ap[2];
        {
            const __bf16* p = &pbuf[wid][l16][0] + kh * 8;
            ap[0] = ld2x8(p,      p + 16);
            ap[1] = ld2x8(p + 32, p + 48);
        }
        // ---- O += P * V  (B-fragments from LDS-resident V^T tile) ----
        #pragma unroll
        for (int jd = 0; jd < 4; ++jd) {
            const char* vr = vt + (jd * 16 + l16) * KV_ROW_B + kh * 16;
            v16bf b0 = ld2x8((const __bf16*)vr,        (const __bf16*)(vr + 32));
            v16bf b1 = ld2x8((const __bf16*)(vr + 64), (const __bf16*)(vr + 96));
            accO[jd] = wmma_bf16(ap[0], b0, accO[jd]);
            accO[jd] = wmma_bf16(ap[1], b1, accO[jd]);
        }
        __syncthreads();
        mbase += 64;                      // next 64 KV columns of the mask
    }

    // ---- finalize: O /= rowsum, write [B,S,D] bf16 for the output projection ----
    #pragma unroll
    for (int jd = 0; jd < 4; ++jd) {
        const int d = jd * 16 + l16;
        #pragma unroll
        for (int r = 0; r < 8; ++r) {
            const int q = q0 + r + kh * 8;
            const float v = accO[jd][r] / lrow[r];
            Ob[((size_t)b * SEQ + q) * DMODEL + h * HEAD_DIM + d] = (__bf16)v;
        }
    }
}

// ---------------- output projection GEMM (fp32 result) ----------------
__global__ __launch_bounds__(128, 1)
void out_gemm_kernel(const __bf16* __restrict__ X, const __bf16* __restrict__ W,
                     const float* __restrict__ bias, float* __restrict__ out) {
    const int lane = threadIdx.x & 31;
    const int wid  = threadIdx.x >> 5;
    const int kh   = lane >> 4;
    const int l16  = lane & 15;
    const int m0 = blockIdx.y * 128 + (wid >> 1) * 64;
    const int n0 = blockIdx.x * 128 + (wid & 1) * 64;

    const __bf16* Abase = X + (size_t)(m0 + l16) * DMODEL + kh * 8;
    const __bf16* Bbase = W + (size_t)(n0 + l16) * DMODEL + kh * 8;

    v8f acc[4][4] = {};
    for (int kb = 0; kb < DMODEL; kb += 32) {
        v16bf afr[4];
        #pragma unroll
        for (int i = 0; i < 4; ++i) {
            const __bf16* p = Abase + i * (16 * DMODEL);
            afr[i] = ld2x8(p, p + 16);
        }
        __builtin_prefetch(Abase + 64, 0, 1);
        #pragma unroll
        for (int j = 0; j < 4; ++j) {
            const __bf16* p = Bbase + j * (16 * DMODEL);
            v16bf bfr = ld2x8(p, p + 16);
            #pragma unroll
            for (int i = 0; i < 4; ++i)
                acc[i][j] = wmma_bf16(afr[i], bfr, acc[i][j]);
        }
        Abase += 32;
        Bbase += 32;
    }
    #pragma unroll
    for (int i = 0; i < 4; ++i) {
        #pragma unroll
        for (int j = 0; j < 4; ++j) {
            const int n = n0 + j * 16 + l16;
            const float bn = bias[n];
            #pragma unroll
            for (int r = 0; r < 8; ++r) {
                const int m = m0 + i * 16 + r + kh * 8;
                out[(size_t)m * DMODEL + n] = acc[i][j][r] + bn;
            }
        }
    }
}

// ---------------- host launcher ----------------
extern "C" void kernel_launch(void* const* d_in, const int* in_sizes, int n_in,
                              void* d_out, int out_size, void* d_ws, size_t ws_size,
                              hipStream_t stream) {
    const float* x  = (const float*)d_in[0];
    const unsigned char* mask = (const unsigned char*)d_in[1];  // jnp bool = 1 byte
    const float* Wq = (const float*)d_in[2];
    const float* bq = (const float*)d_in[3];
    const float* Wk = (const float*)d_in[4];
    const float* bk = (const float*)d_in[5];
    const float* Wv = (const float*)d_in[6];
    const float* bv = (const float*)d_in[7];
    const float* Wo = (const float*)d_in[8];
    const float* bo = (const float*)d_in[9];

    char* ws = (char*)d_ws;
    const size_t MB = 1ull << 20;
    __bf16* xb  = (__bf16*)(ws);             // 16 MB: x in bf16      [M, D]
    __bf16* Wt  = (__bf16*)(ws + 16 * MB);   //  8 MB: 4 transposed weights [N, K]
    __bf16* Qb  = (__bf16*)(ws + 24 * MB);   // 16 MB: Q  [bh, S, dh]
    __bf16* Kb  = (__bf16*)(ws + 40 * MB);   // 16 MB: K  [bh, S, dh]
    __bf16* Vt  = (__bf16*)(ws + 56 * MB);   // 16 MB: V^T [bh, dh, S]
    __bf16* Ob  = (__bf16*)(ws + 72 * MB);   // 16 MB: attention out [B, S, D]

    const int nX = MROWS * DMODEL;           // 8M elements
    cvt_x_kernel<<<nX / 256, 256, 0, stream>>>(x, xb, nX);
    cvt_wt_kernel<<<dim3(DMODEL * DMODEL / 256, 1, 4), 256, 0, stream>>>(Wq, Wk, Wv, Wo, Wt);

    qkv_gemm_kernel<<<dim3(DMODEL / 128, MROWS / 128, 3), 128, 0, stream>>>(
        xb, Wt, bq, bk, bv, Qb, Kb, Vt);

    attn_kernel<<<dim3(SEQ / 64, BATCH * HEADS), 128, 0, stream>>>(Qb, Kb, Vt, mask, Ob);

    out_gemm_kernel<<<dim3(DMODEL / 128, MROWS / 128), 128, 0, stream>>>(
        Ob, Wt + 3ull * DMODEL * DMODEL, bo, (float*)d_out);
}
